// Critic_22016002359854
// MI455X (gfx1250) — matrix operand order repaired and checked by reading, not compiled, not run on hardware
//
#include <hip/hip_runtime.h>

typedef _Float16 half_t;
typedef __attribute__((ext_vector_type(16))) _Float16 v16h;
typedef __attribute__((ext_vector_type(8)))  _Float16 v8h;
typedef __attribute__((ext_vector_type(8)))  float    v8f;

#define N_LINKS 200000
#define N_PAIRS 3200000
#define DD      16
#define ITERS   8

// order-preserving float<->uint key (unsigned compare == float compare)
__device__ __forceinline__ unsigned fenc(float f) {
    unsigned b = __float_as_uint(f);
    return (b & 0x80000000u) ? ~b : (b | 0x80000000u);
}
__device__ __forceinline__ float fdec(unsigned u) {
    unsigned b = (u & 0x80000000u) ? (u ^ 0x80000000u) : ~u;
    return __uint_as_float(b);
}
#define KMAX_INIT 0x00800000u   // fenc(-FLT_MAX)
#define KMIN_INIT 0xFF7FFFFFu   // fenc(+FLT_MAX)

// Branch-free tanh: native V_TANH_F32 if the toolchain exposes it, else
// 1 - 2*rcp(exp2(2*log2e*x)+1) via v_exp_f32 + v_rcp_f32 (saturates to +-1
// automatically for large |x|; no divergent branches).
__device__ __forceinline__ float tanh_fast(float x) {
#if __has_builtin(__builtin_amdgcn_tanhf)
    return __builtin_amdgcn_tanhf(x);
#else
    float e = __builtin_amdgcn_exp2f(x * 2.8853900817779268f);
    return 1.0f - 2.0f * __builtin_amdgcn_rcpf(e + 1.0f);
#endif
}

__device__ __forceinline__ v8f wmma16(v16h a, v16h b, v8f c) {
    return __builtin_amdgcn_wmma_f32_16x16x32_f16(false, a, false, b, (short)0, c, false, false);
}

// ---------------------------------------------------------------------------
// Pack weights into WMMA B-fragment layout.
// Fragment f: 512 halfs, index lane*16 + j.
// Lane L: n = ct*16 + (L&15); k = kc*32 + (j<8 ? (L>>4)*8 + j : 16 + (L>>4)*8 + j-8).  Zero-pad k >= Kact.
// frag ids: [0..3] Wm1(ct)  [4..5] Wm2(kc)  [6..21] Wu1(ct*2+kc, K padded 48->64)
//           [22..37] Wu2(ct*4+kc)  [38..39] Wu3(kc)
// ---------------------------------------------------------------------------
__global__ void pack_weights(const float* __restrict__ Wm1, const float* __restrict__ Wm2,
                             const float* __restrict__ Wu1, const float* __restrict__ Wu2,
                             const float* __restrict__ Wu3, half_t* __restrict__ frags) {
    int f = blockIdx.x;
    int t = threadIdx.x;              // 0..511
    int L = t >> 4, j = t & 15;
    int nl = L & 15, kh = L >> 4;
    const float* W; int Kact, N, ct, kc;
    if (f < 4)       { W = Wm1; Kact = 32;  N = 64;  ct = f;          kc = 0; }
    else if (f < 6)  { W = Wm2; Kact = 64;  N = 16;  ct = 0;          kc = f - 4; }
    else if (f < 22) { W = Wu1; Kact = 48;  N = 128; ct = (f-6) >> 1; kc = (f-6) & 1; }
    else if (f < 38) { W = Wu2; Kact = 128; N = 64;  ct = (f-22) >> 2; kc = (f-22) & 3; }
    else             { W = Wu3; Kact = 64;  N = 16;  ct = 0;          kc = f - 38; }
    int k = kc * 32 + ((j < 8) ? (kh * 8 + j) : (16 + kh * 8 + (j - 8)));
    int n = ct * 16 + nl;
    float v = (k < Kact) ? W[k * N + n] : 0.0f;
    frags[f * 512 + t] = (half_t)v;
}

__global__ void init_ls(const float* __restrict__ x, float* __restrict__ ls) {
    int i = blockIdx.x * blockDim.x + threadIdx.x;
    if (i >= N_LINKS * DD) return;
    int n = i >> 4, f = i & 15;
    ls[i] = (f < 2) ? x[f * N_LINKS + n] : 0.0f;
}

__global__ void init_agg(unsigned* __restrict__ kmax, unsigned* __restrict__ kmin) {
    int i = blockIdx.x * blockDim.x + threadIdx.x;
    if (i >= N_LINKS * DD) return;
    kmax[i] = KMAX_INIT;
    kmin[i] = KMIN_INIT;
}

// ---------------------------------------------------------------------------
// Message phase: one wave per 16-edge tile.
// ---------------------------------------------------------------------------
__global__ __launch_bounds__(256) void msg_kernel(
    const float* __restrict__ ls,
    const int* __restrict__ inc, const int* __restrict__ outc,
    const half_t* __restrict__ frags,
    const float* __restrict__ bm1, const float* __restrict__ bm2,
    unsigned* __restrict__ kmax, unsigned* __restrict__ kmin)
{
    __shared__ half_t stage[8][16 * 64];
    const int lane = threadIdx.x & 31;
    const int wib  = threadIdx.x >> 5;
    const int wave = blockIdx.x * 8 + wib;
    const int nwave = gridDim.x * 8;
    const int nl = lane & 15, hl = lane >> 4, koff = hl * 8;
    half_t* st = stage[wib];

    v16h B1[4], B2[2];
#pragma unroll
    for (int ct = 0; ct < 4; ++ct)
        B1[ct] = *(const v16h*)(frags + ((size_t)ct * 512 + lane * 16));
#pragma unroll
    for (int kc = 0; kc < 2; ++kc)
        B2[kc] = *(const v16h*)(frags + ((size_t)(4 + kc) * 512 + lane * 16));
    float b1v[4];
#pragma unroll
    for (int ct = 0; ct < 4; ++ct) b1v[ct] = bm1[ct * 16 + nl];
    const float b2v = bm2[nl];

    const int NT = N_PAIRS / 16;
    for (int tile = wave; tile < NT; tile += nwave) {
        const int e = tile * 16 + nl;               // A row = lane&15
        const int gi = inc[e], go = outc[e];
        const float4* ip = (const float4*)(ls + (size_t)gi * DD + koff);
        const float4* op = (const float4*)(ls + (size_t)go * DD + koff);
        float4 i0 = ip[0], i1 = ip[1], o0 = op[0], o1 = op[1];
        v16h a;
        a[0] = (half_t)i0.x; a[1] = (half_t)i0.y; a[2] = (half_t)i0.z; a[3] = (half_t)i0.w;
        a[4] = (half_t)i1.x; a[5] = (half_t)i1.y; a[6] = (half_t)i1.z; a[7] = (half_t)i1.w;
        a[8] = (half_t)o0.x; a[9] = (half_t)o0.y; a[10] = (half_t)o0.z; a[11] = (half_t)o0.w;
        a[12] = (half_t)o1.x; a[13] = (half_t)o1.y; a[14] = (half_t)o1.z; a[15] = (half_t)o1.w;

        // layer 1: [16,32] @ [32,64]
#pragma unroll
        for (int ct = 0; ct < 4; ++ct) {
            v8f c = {};
            c = wmma16(a, B1[ct], c);
#pragma unroll
            for (int r = 0; r < 8; ++r) {
                float v = tanh_fast(c[r] + b1v[ct]);
                st[(r + 8 * hl) * 64 + ct * 16 + nl] = (half_t)v;
            }
        }
        __builtin_amdgcn_wave_barrier();

        // layer 2: [16,64] @ [64,16]
        v8f c2 = {};
#pragma unroll
        for (int kc = 0; kc < 2; ++kc) {
            v8h lo = *(const v8h*)(st + nl * 64 + kc * 32 + koff);
            v8h hi = *(const v8h*)(st + nl * 64 + kc * 32 + 16 + koff);
            v16h a2;
#pragma unroll
            for (int j = 0; j < 8; ++j) { a2[j] = lo[j]; a2[8 + j] = hi[j]; }
            c2 = wmma16(a2, B2[kc], c2);
        }
        __builtin_amdgcn_wave_barrier();

        // segment max/min via order-preserving uint keys
#pragma unroll
        for (int r = 0; r < 8; ++r) {
            const int M = r + 8 * hl;
            float v = tanh_fast(c2[r] + b2v);
            unsigned key = fenc(v);
            int seg = outc[tile * 16 + M];
            atomicMax(&kmax[(size_t)seg * DD + nl], key);
            atomicMin(&kmin[(size_t)seg * DD + nl], key);
        }
    }
}

// ---------------------------------------------------------------------------
// Update phase: one wave per 16-node tile.
// ---------------------------------------------------------------------------
__global__ __launch_bounds__(256) void upd_kernel(
    const float* __restrict__ ls,
    const unsigned* __restrict__ kmax, const unsigned* __restrict__ kmin,
    const half_t* __restrict__ frags,
    const float* __restrict__ bu1, const float* __restrict__ bu2,
    const float* __restrict__ bu3,
    float* __restrict__ ls_next)
{
    __shared__ half_t stage1[8][16 * 128];
    __shared__ half_t stage2[8][16 * 64];
    const int lane = threadIdx.x & 31;
    const int wib  = threadIdx.x >> 5;
    const int wave = blockIdx.x * 8 + wib;
    const int nwave = gridDim.x * 8;
    const int nl = lane & 15, hl = lane >> 4, koff = hl * 8;
    half_t* s1 = stage1[wib];
    half_t* s2 = stage2[wib];

    float b1v[8], b2v[4];
#pragma unroll
    for (int ct = 0; ct < 8; ++ct) b1v[ct] = bu1[ct * 16 + nl];
#pragma unroll
    for (int ct = 0; ct < 4; ++ct) b2v[ct] = bu2[ct * 16 + nl];
    const float b3v = bu3[nl];
    v16h BU3[2];
#pragma unroll
    for (int kc = 0; kc < 2; ++kc)
        BU3[kc] = *(const v16h*)(frags + ((size_t)(38 + kc) * 512 + lane * 16));

    const half_t hz = (half_t)0.0f;
    const int NT = N_LINKS / 16;
    for (int tile = wave; tile < NT; tile += nwave) {
        const int node = tile * 16 + nl;
        const float4* lp = (const float4*)(ls + (size_t)node * DD + koff);
        float4 l0 = lp[0], l1 = lp[1];
        const uint4* mp = (const uint4*)(kmax + (size_t)node * DD + koff);
        uint4 x0 = mp[0], x1 = mp[1];
        const uint4* np = (const uint4*)(kmin + (size_t)node * DD + koff);
        uint4 n0 = np[0], n1 = np[1];

        v16h a0, a1;
        a0[0] = (half_t)l0.x; a0[1] = (half_t)l0.y; a0[2] = (half_t)l0.z; a0[3] = (half_t)l0.w;
        a0[4] = (half_t)l1.x; a0[5] = (half_t)l1.y; a0[6] = (half_t)l1.z; a0[7] = (half_t)l1.w;
        a0[8]  = (half_t)fdec(x0.x); a0[9]  = (half_t)fdec(x0.y);
        a0[10] = (half_t)fdec(x0.z); a0[11] = (half_t)fdec(x0.w);
        a0[12] = (half_t)fdec(x1.x); a0[13] = (half_t)fdec(x1.y);
        a0[14] = (half_t)fdec(x1.z); a0[15] = (half_t)fdec(x1.w);
        a1[0] = (half_t)fdec(n0.x); a1[1] = (half_t)fdec(n0.y);
        a1[2] = (half_t)fdec(n0.z); a1[3] = (half_t)fdec(n0.w);
        a1[4] = (half_t)fdec(n1.x); a1[5] = (half_t)fdec(n1.y);
        a1[6] = (half_t)fdec(n1.z); a1[7] = (half_t)fdec(n1.w);
#pragma unroll
        for (int j = 8; j < 16; ++j) a1[j] = hz;   // K 48..63 zero pad

        // layer 1: [16,64] @ [64,128]
        for (int ct = 0; ct < 8; ++ct) {
            v16h w0 = *(const v16h*)(frags + ((size_t)(6 + ct * 2 + 0) * 512 + lane * 16));
            v16h w1 = *(const v16h*)(frags + ((size_t)(6 + ct * 2 + 1) * 512 + lane * 16));
            v8f c = {};
            c = wmma16(a0, w0, c);
            c = wmma16(a1, w1, c);
#pragma unroll
            for (int r = 0; r < 8; ++r) {
                float v = tanh_fast(c[r] + b1v[ct]);
                s1[(r + 8 * hl) * 128 + ct * 16 + nl] = (half_t)v;
            }
        }
        __builtin_amdgcn_wave_barrier();

        // layer 2: [16,128] @ [128,64]
        for (int ct = 0; ct < 4; ++ct) {
            v8f c = {};
#pragma unroll
            for (int kc = 0; kc < 4; ++kc) {
                v8h lo = *(const v8h*)(s1 + nl * 128 + kc * 32 + koff);
                v8h hi = *(const v8h*)(s1 + nl * 128 + kc * 32 + 16 + koff);
                v16h a2;
#pragma unroll
                for (int j = 0; j < 8; ++j) { a2[j] = lo[j]; a2[8 + j] = hi[j]; }
                v16h w = *(const v16h*)(frags + ((size_t)(22 + ct * 4 + kc) * 512 + lane * 16));
                c = wmma16(a2, w, c);
            }
#pragma unroll
            for (int r = 0; r < 8; ++r) {
                float v = tanh_fast(c[r] + b2v[ct]);
                s2[(r + 8 * hl) * 64 + ct * 16 + nl] = (half_t)v;
            }
        }
        __builtin_amdgcn_wave_barrier();

        // layer 3: [16,64] @ [64,16]
        v8f c3 = {};
#pragma unroll
        for (int kc = 0; kc < 2; ++kc) {
            v8h lo = *(const v8h*)(s2 + nl * 64 + kc * 32 + koff);
            v8h hi = *(const v8h*)(s2 + nl * 64 + kc * 32 + 16 + koff);
            v16h a3;
#pragma unroll
            for (int j = 0; j < 8; ++j) { a3[j] = lo[j]; a3[8 + j] = hi[j]; }
            c3 = wmma16(a3, BU3[kc], c3);
        }
#pragma unroll
        for (int r = 0; r < 8; ++r) {
            float v = tanh_fast(c3[r] + b3v);
            ls_next[(size_t)(tile * 16 + r + 8 * hl) * DD + nl] = v;
        }
        __builtin_amdgcn_wave_barrier();
    }
}

// ---------------------------------------------------------------------------
// Readout reduction + tiny MLP
// ---------------------------------------------------------------------------
__global__ void reduce_init(float* sum, float* sumsq, unsigned* rkmax, unsigned* rkmin) {
    int t = threadIdx.x;
    if (t < 16) { sum[t] = 0.0f; sumsq[t] = 0.0f; rkmax[t] = KMAX_INIT; rkmin[t] = KMIN_INIT; }
}

__global__ __launch_bounds__(256) void reduce_ls(
    const float* __restrict__ ls,
    float* __restrict__ sum, float* __restrict__ sumsq,
    unsigned* __restrict__ rkmax, unsigned* __restrict__ rkmin)
{
    __shared__ float ssum[256], ssq[256];
    __shared__ unsigned smax[256], smin[256];
    const int t = threadIdx.x;
    const int col = t & 15, sub = t >> 4;     // [sub 0..15][col 0..15]
    float ac = 0.0f, aq = 0.0f, mx = -3.402823466e38f, mn = 3.402823466e38f;
    for (int node = blockIdx.x * 16 + sub; node < N_LINKS; node += gridDim.x * 16) {
        float v = ls[(size_t)node * DD + col];
        ac += v; aq += v * v;
        mx = fmaxf(mx, v); mn = fminf(mn, v);
    }
    ssum[t] = ac; ssq[t] = aq; smax[t] = fenc(mx); smin[t] = fenc(mn);
    __syncthreads();
    for (int s = 8; s > 0; s >>= 1) {
        if (sub < s) {
            ssum[t] += ssum[t + s * 16];
            ssq[t]  += ssq[t + s * 16];
            smax[t] = (smax[t] > smax[t + s * 16]) ? smax[t] : smax[t + s * 16];
            smin[t] = (smin[t] < smin[t + s * 16]) ? smin[t] : smin[t + s * 16];
        }
        __syncthreads();
    }
    if (sub == 0) {
        atomicAdd(&sum[col], ssum[t]);
        atomicAdd(&sumsq[col], ssq[t]);
        atomicMax(&rkmax[col], smax[t]);
        atomicMin(&rkmin[col], smin[t]);
    }
}

__global__ __launch_bounds__(128) void readout(
    const float* __restrict__ sum, const float* __restrict__ sumsq,
    const unsigned* __restrict__ rkmax, const unsigned* __restrict__ rkmin,
    const float* __restrict__ Wr1, const float* __restrict__ br1,
    const float* __restrict__ Wr2, const float* __restrict__ br2,
    const float* __restrict__ Wr3, const float* __restrict__ br3,
    float* __restrict__ out)
{
    __shared__ float ro[64], h1[128], h2[64];
    const int t = threadIdx.x;
    if (t < 16) {
        float mean = sum[t] / (float)N_LINKS;
        float var  = sumsq[t] / (float)N_LINKS - mean * mean;
        ro[t]      = mean;
        ro[16 + t] = fdec(rkmax[t]);
        ro[32 + t] = fdec(rkmin[t]);
        ro[48 + t] = sqrtf(fmaxf(var, 0.0f));
    }
    __syncthreads();
    {
        float a = br1[t];
        for (int k = 0; k < 64; ++k) a += ro[k] * Wr1[k * 128 + t];
        h1[t] = tanh_fast(a);
    }
    __syncthreads();
    if (t < 64) {
        float a = br2[t];
        for (int k = 0; k < 128; ++k) a += h1[k] * Wr2[k * 64 + t];
        h2[t] = tanh_fast(a);
    }
    __syncthreads();
    if (t == 0) {
        float v = br3[0];
        for (int k = 0; k < 64; ++k) v += h2[k] * Wr3[k];
        out[0] = v;
    }
}

// ---------------------------------------------------------------------------
extern "C" void kernel_launch(void* const* d_in, const int* in_sizes, int n_in,
                              void* d_out, int out_size, void* d_ws, size_t ws_size,
                              hipStream_t stream) {
    const float* x    = (const float*)d_in[0];
    const int*   inc  = (const int*)d_in[1];
    const int*   outc = (const int*)d_in[2];
    const float* Wm1 = (const float*)d_in[3];  const float* bm1 = (const float*)d_in[4];
    const float* Wm2 = (const float*)d_in[5];  const float* bm2 = (const float*)d_in[6];
    const float* Wu1 = (const float*)d_in[7];  const float* bu1 = (const float*)d_in[8];
    const float* Wu2 = (const float*)d_in[9];  const float* bu2 = (const float*)d_in[10];
    const float* Wu3 = (const float*)d_in[11]; const float* bu3 = (const float*)d_in[12];
    const float* Wr1 = (const float*)d_in[13]; const float* br1 = (const float*)d_in[14];
    const float* Wr2 = (const float*)d_in[15]; const float* br2 = (const float*)d_in[16];
    const float* Wr3 = (const float*)d_in[17]; const float* br3 = (const float*)d_in[18];

    char* ws = (char*)d_ws;
    size_t off = 0;
    const size_t LS_BYTES = (size_t)N_LINKS * DD * 4;   // 12.8 MB
    float*    ls0   = (float*)(ws + off);    off += LS_BYTES;
    float*    ls1   = (float*)(ws + off);    off += LS_BYTES;
    unsigned* kmax  = (unsigned*)(ws + off); off += LS_BYTES;
    unsigned* kmin  = (unsigned*)(ws + off); off += LS_BYTES;
    half_t*   frags = (half_t*)(ws + off);   off += 41 * 1024;   // 40 frags * 1KB
    float*    rsum  = (float*)(ws + off);    off += 16 * 4;
    float*    rsq   = (float*)(ws + off);    off += 16 * 4;
    unsigned* rkmax = (unsigned*)(ws + off); off += 16 * 4;
    unsigned* rkmin = (unsigned*)(ws + off); off += 16 * 4;

    pack_weights<<<40, 512, 0, stream>>>(Wm1, Wm2, Wu1, Wu2, Wu3, frags);
    init_ls<<<(N_LINKS * DD + 255) / 256, 256, 0, stream>>>(x, ls0);

    float* cur = ls0;
    float* nxt = ls1;
    for (int it = 0; it < ITERS; ++it) {
        init_agg<<<(N_LINKS * DD + 255) / 256, 256, 0, stream>>>(kmax, kmin);
        msg_kernel<<<1024, 256, 0, stream>>>(cur, inc, outc, frags, bm1, bm2, kmax, kmin);
        upd_kernel<<<512, 256, 0, stream>>>(cur, kmax, kmin, frags, bu1, bu2, bu3, nxt);
        float* tmp = cur; cur = nxt; nxt = tmp;
    }

    reduce_init<<<1, 64, 0, stream>>>(rsum, rsq, rkmax, rkmin);
    reduce_ls<<<512, 256, 0, stream>>>(cur, rsum, rsq, rkmax, rkmin);
    readout<<<1, 128, 0, stream>>>(rsum, rsq, rkmax, rkmin,
                                   Wr1, br1, Wr2, br2, Wr3, br3, (float*)d_out);
}
